// CustomizingAttention_69380901699689
// MI455X (gfx1250) — compile-verified, matrix operand order checked by reading, not run on hardware
//
#include <hip/hip_runtime.h>
#include <cmath>

// ---------------------------------------------------------------------------
// Problem constants (from reference)
// ---------------------------------------------------------------------------
#define B_    16
#define QL_   512
#define VL_   512
#define HID_  1024
#define NH_   4
#define DIM_  256
#define CH_   10

typedef _Float16 f16;
typedef __attribute__((ext_vector_type(8)))  _Float16 v8h;
typedef __attribute__((ext_vector_type(16))) _Float16 v16h;
typedef __attribute__((ext_vector_type(8)))  float    v8f;

// GEMM tiling: block tile 128x128, K-step 32 (one WMMA K per step).
// 256 threads = 8 waves (wave32). Wave grid 4(M) x 2(N): wave tile 32x64.
// Double-buffered LDS: async-prefetch tile i+1 while WMMAs consume tile i.
#define BM 128
#define BN 128
#define BK 32
#define LDSS 40   // f16 elements per LDS row (32 + 8 pad); 80B row stride, 16B aligned

// B-tile source modes
#define BM_F32T 0   // global f32, K-major (weights)  -> transpose+convert into LDS
#define BM_F16N 1   // global f16, N-major (kk for scores) -> async direct copy
#define BM_F16T 2   // global f16, K-major (vv for ctx)    -> transpose into LDS
// Epilogues
#define EPI_BIAS_H16  0  // out_f16 = acc + bias[n] (+ addend[row][n])
#define EPI_SCALE_F32 1  // out_f32 = acc * scale
#define EPI_H16       2  // out_f16 = acc
#define EPI_TANH_F32  3  // out_f32 = tanh(acc + bias[n])

// Load one 16x32 f16 fragment row for this lane from LDS (two 16B ds_load_b128).
// Matches ISA 16-bit A/B layout: lanes 0-15 -> K 0..7 & 16..23, lanes 16-31 -> +8.
__device__ __forceinline__ v16h load_frag(const f16* p) {
    v8h lo = *(const v8h*)p;
    v8h hi = *(const v8h*)(p + 16);
    return __builtin_shufflevector(lo, hi, 0,1,2,3,4,5,6,7,8,9,10,11,12,13,14,15);
}

// Async global->LDS copy, 16B per lane, tracked by ASYNCcnt (gfx1250).
// Generic __shared__ pointers live in the LDS aperture: low 32 bits == LDS
// byte offset (ISA 10.2), so truncation yields the VDST operand directly.
__device__ __forceinline__ void async_copy_b128(void* lds_dst, const void* gsrc) {
    unsigned int       l = (unsigned int)(uintptr_t)lds_dst;
    unsigned long long g = (unsigned long long)(uintptr_t)gsrc;
    asm volatile("global_load_async_to_lds_b128 %0, %1, off"
                 :: "v"(l), "v"(g) : "memory");
}
__device__ __forceinline__ void wait_async0() {
    asm volatile("s_wait_asynccnt 0x0" ::: "memory");
}

template<int BMODE, int EPI, bool SPLITA, bool HAS_ADD>
__global__ __launch_bounds__(256)
void gemm_wmma(const f16* __restrict__ A, int lda, long sAb, long sAh,
               const f16* __restrict__ A2, int lda2, int splitK,
               const void* __restrict__ Bsrc, int ldb, long sBb,
               const float* __restrict__ bias,
               const float* __restrict__ addend, int ldadd,
               void* __restrict__ Cout, int ldc, long sCb, long sCh,
               int K, int NB1, float scale)
{
    __shared__ f16 lsA[2][BM * LDSS];
    __shared__ f16 lsB[2][BN * LDSS];

    const int tid  = threadIdx.x;
    const int lane = tid & 31;
    const int wave = tid >> 5;
    const int wm   = wave & 3;    // 4 waves along M
    const int wn   = wave >> 2;   // 2 waves along N
    const int z    = blockIdx.z;
    const int zb   = z % NB1;
    const int zh   = z / NB1;

    const int m0 = blockIdx.y * BM;
    const int n0 = blockIdx.x * BN;

    const f16* Ab = A + (long)zb * sAb + (long)zh * sAh;

    v8f acc[2][4] = {};

    const int rsel = lane & 15;
    const int koff = (lane >> 4) * 8;

    // ---- tile staging into LDS buffer `buf` for K offset `k0` ----
    auto stage = [&](int buf, int k0) {
        // A tile (128x32 f16): 16B async chunks, coalesced
        #pragma unroll
        for (int i = 0; i < 2; ++i) {
            int chunk = tid + i * 256;          // 0..511
            int r = chunk >> 2;                 // row in tile
            int s = (chunk & 3) * 8;            // k segment
            int gk = k0 + s;
            const f16* src;
            if (SPLITA && gk >= splitK)
                src = A2 + (long)(m0 + r) * lda2 + (gk - splitK);
            else
                src = Ab + (long)(m0 + r) * lda + gk;
            async_copy_b128(&lsA[buf][r * LDSS + s], src);
        }
        // B tile staged N-major (128 rows of 32 K-values)
        if (BMODE == BM_F16N) {
            const f16* Bb = (const f16*)Bsrc + (long)z * sBb;
            #pragma unroll
            for (int i = 0; i < 2; ++i) {
                int chunk = tid + i * 256;
                int r = chunk >> 2;
                int s = (chunk & 3) * 8;
                async_copy_b128(&lsB[buf][r * LDSS + s],
                                Bb + (long)(n0 + r) * ldb + k0 + s);
            }
        } else if (BMODE == BM_F16T) {
            const f16* Bb = (const f16*)Bsrc + (long)z * sBb;
            #pragma unroll
            for (int i = 0; i < 16; ++i) {
                int idx = tid + i * 256;        // 0..4095
                int n = idx & 127;
                int k = idx >> 7;
                lsB[buf][n * LDSS + k] = Bb[(long)(k0 + k) * ldb + n0 + n];
            }
        } else { // BM_F32T: f32 K-major weights, transpose + convert
            const float* Bb = (const float*)Bsrc + (long)z * sBb;
            #pragma unroll
            for (int i = 0; i < 16; ++i) {
                int idx = tid + i * 256;
                int n = idx & 127;
                int k = idx >> 7;
                lsB[buf][n * LDSS + k] = (f16)Bb[(long)(k0 + k) * ldb + n0 + n];
            }
        }
    };

    stage(0, 0);  // prologue: prefetch first tile

    int it = 0;
    for (int k0 = 0; k0 < K; k0 += BK, ++it) {
        const int cur = it & 1;
        wait_async0();        // this wave's async copies into `cur` are done
        __syncthreads();      // all waves' copies visible; prev buffer reusable

        if (k0 + BK < K) stage(cur ^ 1, k0 + BK);   // overlap next-tile fetch

        // Prefetch f32 weight panel two K-steps ahead into GL2.
        if (BMODE == BM_F32T && k0 + 2 * BK < K && tid < 128) {
            const float* Bb = (const float*)Bsrc + (long)z * sBb;
            __builtin_prefetch(Bb + (long)(k0 + 2 * BK + (tid >> 2)) * ldb +
                               n0 + (tid & 3) * 32, 0, 1);
        }

        // ---- fragments + 8 WMMAs ----
        v16h afrag[2];
        #pragma unroll
        for (int i = 0; i < 2; ++i)
            afrag[i] = load_frag(&lsA[cur][(wm * 32 + i * 16 + rsel) * LDSS + koff]);
        v16h bfrag[4];
        #pragma unroll
        for (int j = 0; j < 4; ++j)
            bfrag[j] = load_frag(&lsB[cur][(wn * 64 + j * 16 + rsel) * LDSS + koff]);
        #pragma unroll
        for (int i = 0; i < 2; ++i)
            #pragma unroll
            for (int j = 0; j < 4; ++j)
                acc[i][j] = __builtin_amdgcn_wmma_f32_16x16x32_f16(
                    false, afrag[i], false, bfrag[j], (short)0, acc[i][j],
                    false, false);
    }

    // ---- epilogue. C/D layout: lane<16 -> N=lane, M=v (VGPR v); lane>=16 -> M=8+v.
    const long coff = (long)zb * sCb + (long)zh * sCh;
    const int row_lo = (lane >> 4) * 8;

    float bcol[4];
    if (EPI == EPI_BIAS_H16 || EPI == EPI_TANH_F32) {
        #pragma unroll
        for (int j = 0; j < 4; ++j)
            bcol[j] = bias[n0 + wn * 64 + j * 16 + rsel];
    }

    #pragma unroll
    for (int i = 0; i < 2; ++i) {
        #pragma unroll
        for (int j = 0; j < 4; ++j) {
            #pragma unroll
            for (int v = 0; v < 8; ++v) {
                int grow = m0 + wm * 32 + i * 16 + row_lo + v;
                int gcol = n0 + wn * 64 + j * 16 + rsel;
                long cidx = coff + (long)grow * ldc + gcol;
                float val = acc[i][j][v];
                if (EPI == EPI_BIAS_H16) {
                    val += bcol[j];
                    if (HAS_ADD) val += addend[(long)grow * ldadd + gcol];
                    ((f16*)Cout)[cidx] = (f16)val;
                } else if (EPI == EPI_SCALE_F32) {
                    ((float*)Cout)[cidx] = val * scale;
                } else if (EPI == EPI_H16) {
                    ((f16*)Cout)[cidx] = (f16)val;
                } else { // EPI_TANH_F32
                    ((float*)Cout)[cidx] = tanhf(val + bcol[j]);
                }
            }
        }
    }
}

// ---------------------------------------------------------------------------
// f32 -> f16 convert (grid-stride)
// ---------------------------------------------------------------------------
__global__ void convert_kernel(const float* __restrict__ in,
                               f16* __restrict__ out, long n)
{
    long i = (long)blockIdx.x * blockDim.x + threadIdx.x;
    long stride = (long)gridDim.x * blockDim.x;
    for (; i < n; i += stride) out[i] = (f16)in[i];
}

// ---------------------------------------------------------------------------
// loc = (conv1d(prev_attn) + conv_b) @ Wloc, scattered to (B, VL, HID) layout
// blockIdx.x = v, blockIdx.y = n (= b*NH + h), threadIdx.x = d
// ---------------------------------------------------------------------------
__global__ __launch_bounds__(256)
void loc_kernel(const float* __restrict__ prev, const float* __restrict__ cw,
                const float* __restrict__ cb, const float* __restrict__ wloc,
                float* __restrict__ loc)
{
    int v = blockIdx.x;
    int n = blockIdx.y;
    int d = threadIdx.x;
    int b = n >> 2;      // NH_ == 4
    int h = n & 3;
    float pm1 = (v > 0)       ? prev[(long)n * VL_ + v - 1] : 0.f;
    float p0  =                 prev[(long)n * VL_ + v];
    float pp1 = (v < VL_ - 1) ? prev[(long)n * VL_ + v + 1] : 0.f;
    float acc = 0.f;
    #pragma unroll
    for (int c = 0; c < CH_; ++c) {
        float conv = pm1 * cw[c * 3 + 0] + p0 * cw[c * 3 + 1] +
                     pp1 * cw[c * 3 + 2] + cb[c];
        acc += conv * wloc[c * DIM_ + d];
    }
    loc[((long)(b * VL_ + v)) * HID_ + h * DIM_ + d] = acc;
}

// ---------------------------------------------------------------------------
// Row softmax over VL=512; writes f32 attn (output #2) and f16 attn for ctx.
// blockIdx.x = q row, blockIdx.y = batch n, 256 threads, 2 elems/thread.
// ---------------------------------------------------------------------------
__global__ __launch_bounds__(256)
void softmax_kernel(const float* __restrict__ sc, float* __restrict__ attn_f32,
                    f16* __restrict__ attn_f16)
{
    __shared__ float redmax[8];
    __shared__ float redsum[8];
    int q = blockIdx.x, z = blockIdx.y, tid = threadIdx.x;
    long base = ((long)z * QL_ + q) * VL_;
    float v0 = sc[base + tid];
    float v1 = sc[base + tid + 256];
    float m = fmaxf(v0, v1);
    #pragma unroll
    for (int off = 16; off > 0; off >>= 1) m = fmaxf(m, __shfl_xor(m, off, 32));
    if ((tid & 31) == 0) redmax[tid >> 5] = m;
    __syncthreads();
    float M = redmax[0];
    #pragma unroll
    for (int i = 1; i < 8; ++i) M = fmaxf(M, redmax[i]);
    float e0 = __expf(v0 - M), e1 = __expf(v1 - M);
    float s = e0 + e1;
    #pragma unroll
    for (int off = 16; off > 0; off >>= 1) s += __shfl_xor(s, off, 32);
    if ((tid & 31) == 0) redsum[tid >> 5] = s;
    __syncthreads();
    float S = 0.f;
    #pragma unroll
    for (int i = 0; i < 8; ++i) S += redsum[i];
    float inv = 1.0f / S;
    float a0 = e0 * inv, a1 = e1 * inv;
    attn_f32[base + tid]       = a0;
    attn_f32[base + tid + 256] = a1;
    attn_f16[base + tid]       = (f16)a0;
    attn_f16[base + tid + 256] = (f16)a1;
}

// ---------------------------------------------------------------------------
// Host-side orchestration
// ---------------------------------------------------------------------------
extern "C" void kernel_launch(void* const* d_in, const int* in_sizes, int n_in,
                              void* d_out, int out_size, void* d_ws, size_t ws_size,
                              hipStream_t stream)
{
    (void)in_sizes; (void)n_in; (void)out_size; (void)ws_size;

    const float* query  = (const float*)d_in[0];
    const float* value  = (const float*)d_in[1];
    const float* prev   = (const float*)d_in[2];
    const float* conv_w = (const float*)d_in[3];
    const float* conv_b = (const float*)d_in[4];
    const float* Wq     = (const float*)d_in[5];
    const float* bq     = (const float*)d_in[6];
    const float* Wv     = (const float*)d_in[7];
    const float* Wloc   = (const float*)d_in[8];
    const float* biasv  = (const float*)d_in[9];
    const float* Wout   = (const float*)d_in[10];
    const float* bout   = (const float*)d_in[11];
    const float* swq    = (const float*)d_in[12];
    const float* sbq    = (const float*)d_in[13];
    const float* swk    = (const float*)d_in[14];
    const float* sbk    = (const float*)d_in[15];
    const float* swv    = (const float*)d_in[16];
    const float* sbv    = (const float*)d_in[17];

    float* out      = (float*)d_out;                         // (B, QL, HID) f32
    float* attn_out = out + (size_t)B_ * QL_ * HID_;         // (NH*B, QL, VL) f32

    // Workspace layout (256B-aligned slots)
    char* ws = (char*)d_ws;
    size_t off = 0;
    auto alloc = [&](size_t bytes) {
        size_t r = off; off += (bytes + 255) & ~(size_t)255; return r;
    };
    const size_t NQH = (size_t)B_ * QL_ * HID_;              // 8,388,608
    const size_t NHB = (size_t)NH_ * B_;                     // 64
    const size_t NQD = NHB * QL_ * DIM_;                     // 8,388,608
    const size_t NQV = NHB * QL_ * VL_;                      // 16,777,216

    f16*   qry16  = (f16*)(ws + alloc(NQH * 2));
    f16*   val16  = (f16*)(ws + alloc(NQH * 2));
    float* locf   = (float*)(ws + alloc(NQH * 4));
    f16*   q16    = (f16*)(ws + alloc(NQH * 2));
    f16*   v16    = (f16*)(ws + alloc(NQH * 2));
    f16*   qq16   = (f16*)(ws + alloc(NQD * 2));
    f16*   kk16   = (f16*)(ws + alloc(NQD * 2));
    f16*   vv16   = (f16*)(ws + alloc(NQD * 2));
    float* scf    = (float*)(ws + alloc(NQV * 4));
    f16*   attn16 = (f16*)(ws + alloc(NQV * 2));
    f16*   ctx16  = (f16*)(ws + alloc(NQH * 2));

    // Batch strides (elements)
    const long sQHb = (long)QL_ * HID_;   // per-b stride into (B,QL,HID)
    const long sQDb = (long)QL_ * DIM_;   // per-z stride into (NHB,QL,DIM)
    const long sQVb = (long)QL_ * VL_;    // per-z stride into (NHB,QL,VL)

    // 0) converts + loc
    convert_kernel<<<2048, 256, 0, stream>>>(query, qry16, (long)NQH);
    convert_kernel<<<2048, 256, 0, stream>>>(value, val16, (long)NQH);
    loc_kernel<<<dim3(VL_, NH_ * B_), 256, 0, stream>>>(prev, conv_w, conv_b, Wloc, locf);

    // 1) q = query @ Wq + bq          (8192 x 1024 x 1024)
    gemm_wmma<BM_F32T, EPI_BIAS_H16, false, false><<<dim3(8, 64, 1), 256, 0, stream>>>(
        qry16, HID_, 0, 0, nullptr, 0, 0,
        Wq, HID_, 0, bq, nullptr, 0,
        q16, HID_, 0, 0, HID_, 1, 1.f);

    // 2) v = value @ Wv + loc + bias  (8192 x 1024 x 1024)
    gemm_wmma<BM_F32T, EPI_BIAS_H16, false, true><<<dim3(8, 64, 1), 256, 0, stream>>>(
        val16, HID_, 0, 0, nullptr, 0, 0,
        Wv, HID_, 0, biasv, locf, HID_,
        v16, HID_, 0, 0, HID_, 1, 1.f);

    // 3) head projections: z = h*B + b; A offset = b*QL*HID + h*DIM
    gemm_wmma<BM_F32T, EPI_BIAS_H16, false, false><<<dim3(2, 4, 64), 256, 0, stream>>>(
        q16, HID_, sQHb, DIM_, nullptr, 0, 0,
        swq, DIM_, 0, sbq, nullptr, 0,
        qq16, DIM_, sQDb, sQDb * B_, DIM_, B_, 1.f);
    gemm_wmma<BM_F32T, EPI_BIAS_H16, false, false><<<dim3(2, 4, 64), 256, 0, stream>>>(
        v16, HID_, sQHb, DIM_, nullptr, 0, 0,
        swk, DIM_, 0, sbk, nullptr, 0,
        kk16, DIM_, sQDb, sQDb * B_, DIM_, B_, 1.f);
    gemm_wmma<BM_F32T, EPI_BIAS_H16, false, false><<<dim3(2, 4, 64), 256, 0, stream>>>(
        v16, HID_, sQHb, DIM_, nullptr, 0, 0,
        swv, DIM_, 0, sbv, nullptr, 0,
        vv16, DIM_, sQDb, sQDb * B_, DIM_, B_, 1.f);

    // 4) scores = (qq @ kk^T) / sqrt(DIM); kk rows are already N-major
    gemm_wmma<BM_F16N, EPI_SCALE_F32, false, false><<<dim3(4, 4, 64), 256, 0, stream>>>(
        qq16, DIM_, sQDb, sQDb * B_, nullptr, 0, 0,
        kk16, DIM_, sQDb, nullptr, nullptr, 0,
        scf, VL_, sQVb, sQVb * B_, DIM_, B_, 0.0625f);

    // 5) softmax -> attn (f32 output #2) + f16 copy
    softmax_kernel<<<dim3(QL_, NH_ * B_), 256, 0, stream>>>(scf, attn_out, attn16);

    // 6) ctx = attn @ vv, stored merged-head: out offset = b*QL*HID + h*DIM
    gemm_wmma<BM_F16T, EPI_H16, false, false><<<dim3(2, 4, 64), 256, 0, stream>>>(
        attn16, VL_, sQVb, sQVb * B_, nullptr, 0, 0,
        vv16, DIM_, sQDb, nullptr, nullptr, 0,
        ctx16, HID_, sQHb, DIM_, VL_, B_, 1.f);

    // 7) out = tanh([ctx | query] @ Wout + bout)   (8192 x 1024, K = 2048 split)
    gemm_wmma<BM_F32T, EPI_TANH_F32, true, false><<<dim3(8, 64, 1), 256, 0, stream>>>(
        ctx16, HID_, 0, 0, qry16, HID_, HID_,
        Wout, HID_, 0, bout, nullptr, 0,
        out, HID_, 0, 0, 2 * HID_, 1, 1.f);
}
